// DynamicMicroserviceModel_91164975825028
// MI455X (gfx1250) — matrix operand order repaired and checked by reading, not compiled, not run on hardware
//
#include <hip/hip_runtime.h>

typedef __bf16 bf16;
typedef __attribute__((ext_vector_type(16))) __bf16 v16bf;
typedef __attribute__((ext_vector_type(8)))  float  v8f;
typedef __attribute__((ext_vector_type(8)))  unsigned v8u;

#define HEADS 4
#define HID   128

// ---------------------------------------------------------------------------
// helpers
// ---------------------------------------------------------------------------
__device__ __forceinline__ float sigmoidf_(float x) { return 1.0f / (1.0f + __expf(-x)); }
__device__ __forceinline__ unsigned f2ord(float f) {
    unsigned u = __float_as_uint(f);
    return (u & 0x80000000u) ? ~u : (u | 0x80000000u);
}
__device__ __forceinline__ float ord2f(unsigned u) {
    unsigned v = (u & 0x80000000u) ? (u & 0x7FFFFFFFu) : ~u;
    return __uint_as_float(v);
}

// ---------------------------------------------------------------------------
// generic fills / casts
// ---------------------------------------------------------------------------
__global__ void fill_u32(unsigned* __restrict__ p, unsigned v, long n) {
    long i = (long)blockIdx.x * blockDim.x + threadIdx.x;
    long stride = (long)gridDim.x * blockDim.x;
    for (; i < n; i += stride) p[i] = v;
}

__global__ void cast_f32_bf16(const float* __restrict__ s, bf16* __restrict__ d, long n) {
    long i = (long)blockIdx.x * blockDim.x + threadIdx.x;
    if (i < n) d[i] = (bf16)s[i];
}

__global__ void transpose_cast(const float* __restrict__ s, bf16* __restrict__ d, int R, int C) {
    long i = (long)blockIdx.x * blockDim.x + threadIdx.x;
    if (i >= (long)R * C) return;
    int r = (int)(i / C), c = (int)(i % C);
    d[(size_t)c * R + r] = (bf16)s[i];
}

// build [N,96] bf16 = concat(x_t [N,64], emb [N,32])
__global__ void concat_cast(const float* __restrict__ x, const float* __restrict__ emb,
                            bf16* __restrict__ d, int N) {
    long i = (long)blockIdx.x * blockDim.x + threadIdx.x;
    if (i >= (long)N * 96) return;
    int n = (int)(i / 96), j = (int)(i % 96);
    float v = (j < 64) ? x[(size_t)n * 64 + j] : emb[(size_t)n * 32 + (j - 64)];
    d[i] = (bf16)v;
}

// ---------------------------------------------------------------------------
// Tiled WMMA GEMM: C[M,N] = A[M,K](bf16) * B[K,N](bf16) (+bias | +=C)
// 256 threads = 8 waves (2 x 4); tile 64x128, BK=32; each wave: 32x32 output
// = 2 A-frags x 2 B-frags -> 4 v_wmma_f32_16x16x32_bf16 per K-step.
// B staged pair-packed: BsU[k/2][n] dword = (B[k][n], B[k+1][n]) -> fragment
// dword r == BsU[hl*8+r][n] directly (contiguous k = hl*16+s, same order as A), conflict-free.
// ---------------------------------------------------------------------------
#define BM 64
#define BN 128
#define BK 32

__global__ void __launch_bounds__(256)
gemm_bf16_wmma(const bf16* __restrict__ A, const bf16* __restrict__ B,
               const float* __restrict__ bias, float* __restrict__ C,
               int M, int K, int N, int accum)
{
    __shared__ bf16     As[BM][BK + 8];       // row-major A tile, padded
    __shared__ unsigned BsU[BK / 2][BN];      // pair-packed B tile

    const int tid  = threadIdx.x;
    const int lane = tid & 31;
    const int wid  = tid >> 5;
    const int wm   = wid & 1;          // 0..1 : 32-row sub-block
    const int wn   = wid >> 1;         // 0..3 : 32-col sub-block
    const int m0   = blockIdx.x * BM;
    const int n0   = blockIdx.y * BN;
    const int hl   = lane >> 4;
    const int l15  = lane & 15;

    // staging coordinates
    const int ar  = tid >> 2;          // 0..63  (A row)
    const int ac  = (tid & 3) * 8;     // 0,8,16,24
    const int bkp = tid >> 4;          // 0..15  (B k-pair)
    const int bnt = (tid & 15) * 8;    // 0..120

    v8f acc[2][2] = {};

    for (int k0 = 0; k0 < K; k0 += BK) {
        if (k0 + BK < K) {   // prefetch next tiles (global_prefetch_b8)
            int pr = (m0 + ar < M) ? (m0 + ar) : m0;
            __builtin_prefetch(A + (size_t)pr * K + k0 + BK + ac, 0, 1);
            __builtin_prefetch(B + (size_t)(k0 + BK + 2 * bkp) * N + n0 + bnt, 0, 1);
        }
        // stage A (contiguous b128)
        {
            int gr = m0 + ar;
            uint4 v = make_uint4(0u, 0u, 0u, 0u);
            if (gr < M) v = *(const uint4*)(A + (size_t)gr * K + k0 + ac);
            *(uint4*)&As[ar][ac] = v;
        }
        // stage B pair-packed (two row reads, VALU pack, contiguous b128 stores)
        {
            const bf16* prow0 = B + (size_t)(k0 + 2 * bkp) * N + n0 + bnt;
            uint4 lo = *(const uint4*)prow0;
            uint4 hi = *(const uint4*)(prow0 + N);
            uint4 q0, q1;
            q0.x = (lo.x & 0xFFFFu) | (hi.x << 16);
            q0.y = (lo.x >> 16)     | (hi.x & 0xFFFF0000u);
            q0.z = (lo.y & 0xFFFFu) | (hi.y << 16);
            q0.w = (lo.y >> 16)     | (hi.y & 0xFFFF0000u);
            q1.x = (lo.z & 0xFFFFu) | (hi.z << 16);
            q1.y = (lo.z >> 16)     | (hi.z & 0xFFFF0000u);
            q1.z = (lo.w & 0xFFFFu) | (hi.w << 16);
            q1.w = (lo.w >> 16)     | (hi.w & 0xFFFF0000u);
            *(uint4*)&BsU[bkp][bnt]     = q0;
            *(uint4*)&BsU[bkp][bnt + 4] = q1;
        }
        __syncthreads();

        // fragments
        v16bf a0, a1, b0, b1;
        {
            const bf16* r0 = &As[wm * 32 + l15][0];
            const bf16* r1 = &As[wm * 32 + 16 + l15][0];
#pragma unroll
            for (int e = 0; e < 8; ++e) {       // a[e] -> K = hl*16+e ; a[8+e] -> K = hl*16+8+e
                a0[e]     = r0[hl * 16 + e];
                a0[8 + e] = r0[hl * 16 + 8 + e];
                a1[e]     = r1[hl * 16 + e];
                a1[8 + e] = r1[hl * 16 + 8 + e];
            }
            int nA = wn * 32 + l15;
            v8u u0, u1;
#pragma unroll
            for (int r = 0; r < 8; ++r) {       // frag dword r = (K=hl*16+2r, +1) at col n
                u0[r] = BsU[hl * 8 + r][nA];
                u1[r] = BsU[hl * 8 + r][nA + 16];
            }
            b0 = __builtin_bit_cast(v16bf, u0);
            b1 = __builtin_bit_cast(v16bf, u1);
        }
        acc[0][0] = __builtin_amdgcn_wmma_f32_16x16x32_bf16(false, a0, false, b0, (short)0, acc[0][0], false, false);
        acc[0][1] = __builtin_amdgcn_wmma_f32_16x16x32_bf16(false, a0, false, b1, (short)0, acc[0][1], false, false);
        acc[1][0] = __builtin_amdgcn_wmma_f32_16x16x32_bf16(false, a1, false, b0, (short)0, acc[1][0], false, false);
        acc[1][1] = __builtin_amdgcn_wmma_f32_16x16x32_bf16(false, a1, false, b1, (short)0, acc[1][1], false, false);
        __syncthreads();
    }

    // store (C layout: lane n = l15, VGPR r -> row half*8+r)
#pragma unroll
    for (int i = 0; i < 2; ++i) {
#pragma unroll
        for (int j = 0; j < 2; ++j) {
            int col = n0 + wn * 32 + j * 16 + l15;
            float bv = bias ? bias[col] : 0.0f;
#pragma unroll
            for (int r = 0; r < 8; ++r) {
                int row = m0 + wm * 32 + i * 16 + hl * 8 + r;
                if (row < M) {
                    size_t idx = (size_t)row * N + col;
                    float base = accum ? C[idx] : bv;
                    C[idx] = base + acc[i][j][r];
                }
            }
        }
    }
}

// ---------------------------------------------------------------------------
// GATv2 edge kernels
// ---------------------------------------------------------------------------
__global__ void __launch_bounds__(256)
edge_scores(const float* __restrict__ xl, const float* __restrict__ xr,
            const int* __restrict__ srcA, const int* __restrict__ dstA,
            const float* __restrict__ att, float* __restrict__ ev,
            unsigned* __restrict__ emaxu, int Etot, int Eorig)
{
    int gw   = (int)(((long)blockIdx.x * blockDim.x + threadIdx.x) >> 5);
    int lane = threadIdx.x & 31;
    if (gw >= Etot * HEADS) return;
    int e = gw >> 2, h = gw & 3;
    int s, d;
    if (e < Eorig) { s = srcA[e]; d = dstA[e]; } else { s = d = e - Eorig; }
    const float* pl = xl + (size_t)s * (HEADS * HID) + h * HID;
    const float* pr = xr + (size_t)d * (HEADS * HID) + h * HID;
    const float* pa = att + h * HID;
    float acc = 0.0f;
#pragma unroll
    for (int i = 0; i < 4; ++i) {
        int c = lane + 32 * i;
        float v = pl[c] + pr[c];
        float lr = v > 0.0f ? v : 0.2f * v;
        acc += lr * pa[c];
    }
#pragma unroll
    for (int off = 16; off > 0; off >>= 1) acc += __shfl_xor(acc, off, 32);
    if (lane == 0) {
        ev[gw] = acc;
        atomicMax(&emaxu[(size_t)d * HEADS + h], f2ord(acc));
    }
}

__global__ void edge_exp(float* __restrict__ ev, const unsigned* __restrict__ emaxu,
                         float* __restrict__ denom,
                         const int* __restrict__ dstA, int Etot, int Eorig)
{
    int idx = (int)((long)blockIdx.x * blockDim.x + threadIdx.x);
    if (idx >= Etot * HEADS) return;
    int e = idx >> 2, h = idx & 3;
    int d = (e < Eorig) ? dstA[e] : (e - Eorig);
    float m = ord2f(emaxu[(size_t)d * HEADS + h]);
    float x = __expf(ev[idx] - m);
    ev[idx] = x;
    atomicAdd(&denom[(size_t)d * HEADS + h], x);
}

__global__ void __launch_bounds__(256)
edge_aggr(const float* __restrict__ xl, const float* __restrict__ ev,
          const float* __restrict__ denom,
          const int* __restrict__ srcA, const int* __restrict__ dstA,
          float* __restrict__ agg, int Etot, int Eorig)
{
    int gw   = (int)(((long)blockIdx.x * blockDim.x + threadIdx.x) >> 5);
    int lane = threadIdx.x & 31;
    if (gw >= Etot * HEADS) return;
    int e = gw >> 2, h = gw & 3;
    int s, d;
    if (e < Eorig) { s = srcA[e]; d = dstA[e]; } else { s = d = e - Eorig; }
    float alpha = ev[gw] / denom[(size_t)d * HEADS + h];
    const float* pl = xl + (size_t)s * (HEADS * HID) + h * HID;
    float* pd = agg + (size_t)d * (HEADS * HID) + h * HID;
#pragma unroll
    for (int i = 0; i < 4; ++i) {
        int c = lane + 32 * i;
        atomicAdd(&pd[c], alpha * pl[c]);
    }
}

// per-node: mean over heads + gat bias + residual, then LayerNorm
__global__ void __launch_bounds__(128)
node_ln(const float* __restrict__ agg, const float* __restrict__ gb,
        const float* __restrict__ res, const float* __restrict__ g,
        const float* __restrict__ b, float* __restrict__ outf,
        bf16* __restrict__ outb)
{
    int n = blockIdx.x, j = threadIdx.x;
    size_t base = (size_t)n * (HEADS * HID);
    float v = 0.25f * (agg[base + j] + agg[base + HID + j] +
                       agg[base + 2 * HID + j] + agg[base + 3 * HID + j]) +
              gb[j] + res[(size_t)n * HID + j];
    __shared__ float red[HID];
    red[j] = v; __syncthreads();
    for (int s = 64; s > 0; s >>= 1) { if (j < s) red[j] += red[j + s]; __syncthreads(); }
    float mu = red[0] / (float)HID; __syncthreads();
    float dv = v - mu;
    red[j] = dv * dv; __syncthreads();
    for (int s = 64; s > 0; s >>= 1) { if (j < s) red[j] += red[j + s]; __syncthreads(); }
    float var = red[0] / (float)HID;
    float o = dv * rsqrtf(var + 1e-5f) * g[j] + b[j];
    if (outf) outf[(size_t)n * HID + j] = o;
    if (outb) outb[(size_t)n * HID + j] = (bf16)o;
}

// ---------------------------------------------------------------------------
// LSTM gates (PyTorch order i,f,g,o); z = x@Wih^T + h@Whh^T already in z
// ---------------------------------------------------------------------------
__global__ void lstm_gate(const float* __restrict__ z, const float* __restrict__ bih,
                          const float* __restrict__ bhh, float* __restrict__ hs,
                          float* __restrict__ cs, bf16* __restrict__ hsb,
                          float* __restrict__ y, int N)
{
    long i = (long)blockIdx.x * blockDim.x + threadIdx.x;
    if (i >= (long)N * HID) return;
    int n = (int)(i >> 7), j = (int)(i & (HID - 1));
    const float* zr = z + (size_t)n * (4 * HID);
    float ig = zr[j]           + bih[j]           + bhh[j];
    float fg = zr[HID + j]     + bih[HID + j]     + bhh[HID + j];
    float gg = zr[2 * HID + j] + bih[2 * HID + j] + bhh[2 * HID + j];
    float og = zr[3 * HID + j] + bih[3 * HID + j] + bhh[3 * HID + j];
    float c = sigmoidf_(fg) * cs[i] + sigmoidf_(ig) * tanhf(gg);
    float h = sigmoidf_(og) * tanhf(c);
    cs[i] = c; hs[i] = h; hsb[i] = (bf16)h; y[i] = h;
}

// ---------------------------------------------------------------------------
// attention pooling over T + fc1(relu) + fc2 -> out[n]
// ---------------------------------------------------------------------------
__global__ void __launch_bounds__(256)
attn_fc(const float* __restrict__ yf, const float* __restrict__ yb,
        const float* __restrict__ aW, const float* __restrict__ ab,
        const float* __restrict__ f1W, const float* __restrict__ f1b,
        const float* __restrict__ f2W, const float* __restrict__ f2b,
        float* __restrict__ out, int N, int T)
{
    int n = blockIdx.x, j = threadIdx.x;
    __shared__ float red[256];
    __shared__ float logit[8];
    __shared__ float wsm[8];
    __shared__ float ctx[256];
    __shared__ float h64[64];
    float ov[8];
    for (int t = 0; t < T; ++t) {
        float v = (j < HID) ? yf[((size_t)t * N + n) * HID + j]
                            : yb[((size_t)t * N + n) * HID + (j - HID)];
        ov[t] = v;
        red[j] = v * aW[j];
        __syncthreads();
        for (int s = 128; s > 0; s >>= 1) { if (j < s) red[j] += red[j + s]; __syncthreads(); }
        if (j == 0) logit[t] = red[0] + ab[0];
        __syncthreads();
    }
    if (j == 0) {
        float mx = logit[0];
        for (int t = 1; t < T; ++t) mx = fmaxf(mx, logit[t]);
        float s = 0.0f;
        for (int t = 0; t < T; ++t) { wsm[t] = __expf(logit[t] - mx); s += wsm[t]; }
        for (int t = 0; t < T; ++t) wsm[t] /= s;
    }
    __syncthreads();
    float c = 0.0f;
    for (int t = 0; t < T; ++t) c += wsm[t] * ov[t];
    ctx[j] = c; __syncthreads();
    if (j < 64) {
        float a = f1b[j];
        for (int q = 0; q < 256; ++q) a += ctx[q] * f1W[q * 64 + j];
        h64[j] = fmaxf(a, 0.0f);
    }
    __syncthreads();
    if (j == 0) {
        float s = f2b[0];
        for (int k = 0; k < 64; ++k) s += h64[k] * f2W[k];
        out[n] = s;
    }
}

// ---------------------------------------------------------------------------
// host launcher
// ---------------------------------------------------------------------------
static inline char* ws_take(char*& cur, size_t bytes) {
    char* p = cur;
    cur += (bytes + 255) & ~(size_t)255;
    return p;
}

extern "C" void kernel_launch(void* const* d_in, const int* in_sizes, int n_in,
                              void* d_out, int out_size, void* d_ws, size_t ws_size,
                              hipStream_t stream)
{
    (void)n_in; (void)out_size; (void)ws_size;

    const float* x      = (const float*)d_in[0];
    const int*   ei     = (const int*)d_in[1];
    const float* emb    = (const float*)d_in[2];
    const float* projW  = (const float*)d_in[3];
    const float* projb  = (const float*)d_in[4];
    const float* g1Wl   = (const float*)d_in[5];
    const float* g1bl   = (const float*)d_in[6];
    const float* g1Wr   = (const float*)d_in[7];
    const float* g1br   = (const float*)d_in[8];
    const float* g1att  = (const float*)d_in[9];
    const float* g1bias = (const float*)d_in[10];
    const float* g2Wl   = (const float*)d_in[11];
    const float* g2bl   = (const float*)d_in[12];
    const float* g2Wr   = (const float*)d_in[13];
    const float* g2br   = (const float*)d_in[14];
    const float* g2att  = (const float*)d_in[15];
    const float* g2bias = (const float*)d_in[16];
    const float* ln1g   = (const float*)d_in[17];
    const float* ln1b   = (const float*)d_in[18];
    const float* ln2g   = (const float*)d_in[19];
    const float* ln2b   = (const float*)d_in[20];
    const float* Wih_f  = (const float*)d_in[21];
    const float* Whh_f  = (const float*)d_in[22];
    const float* bih_f  = (const float*)d_in[23];
    const float* bhh_f  = (const float*)d_in[24];
    const float* Wih_b  = (const float*)d_in[25];
    const float* Whh_b  = (const float*)d_in[26];
    const float* bih_b  = (const float*)d_in[27];
    const float* bhh_b  = (const float*)d_in[28];
    const float* attnW  = (const float*)d_in[29];
    const float* attnb  = (const float*)d_in[30];
    const float* fc1W   = (const float*)d_in[31];
    const float* fc1b   = (const float*)d_in[32];
    const float* fc2W   = (const float*)d_in[33];
    const float* fc2b   = (const float*)d_in[34];

    const int N     = in_sizes[2] / 32;          // EMB = 32
    const int Eorig = in_sizes[1] / 2;
    const int T     = in_sizes[0] / (N * 64);    // IN_CH = 64
    const int Etot  = Eorig + N;
    const int* srcA = ei;
    const int* dstA = ei + Eorig;

    // ---- workspace layout (persistent + aliased scratch) ----
    char* cur = (char*)d_ws;
    bf16* projWb = (bf16*)ws_take(cur, (size_t)96 * 128 * 2);
    bf16* g1Wlb  = (bf16*)ws_take(cur, (size_t)128 * 512 * 2);
    bf16* g1Wrb  = (bf16*)ws_take(cur, (size_t)128 * 512 * 2);
    bf16* g2Wlb  = (bf16*)ws_take(cur, (size_t)128 * 512 * 2);
    bf16* g2Wrb  = (bf16*)ws_take(cur, (size_t)128 * 512 * 2);
    bf16* WihTf  = (bf16*)ws_take(cur, (size_t)128 * 512 * 2);
    bf16* WhhTf  = (bf16*)ws_take(cur, (size_t)128 * 512 * 2);
    bf16* WihTb  = (bf16*)ws_take(cur, (size_t)128 * 512 * 2);
    bf16* WhhTb  = (bf16*)ws_take(cur, (size_t)128 * 512 * 2);
    bf16* seqb   = (bf16*)ws_take(cur, (size_t)T * N * HID * 2);
    float* yfub  = (float*)ws_take(cur, (size_t)T * N * HID * 4);
    float* ybub  = (float*)ws_take(cur, (size_t)T * N * HID * 4);
    char* scratch = cur;
    // GAT-phase scratch
    char* gc = scratch;
    bf16*     A96b  = (bf16*)ws_take(gc, (size_t)N * 96 * 2);
    float*    xt    = (float*)ws_take(gc, (size_t)N * HID * 4);
    float*    hbuf  = (float*)ws_take(gc, (size_t)N * HID * 4);
    bf16*     nodeb = (bf16*)ws_take(gc, (size_t)N * HID * 2);
    float*    xl    = (float*)ws_take(gc, (size_t)N * HEADS * HID * 4);
    float*    xr    = (float*)ws_take(gc, (size_t)N * HEADS * HID * 4);
    float*    agg   = (float*)ws_take(gc, (size_t)N * HEADS * HID * 4);
    float*    ev    = (float*)ws_take(gc, (size_t)Etot * HEADS * 4);
    unsigned* emaxu = (unsigned*)ws_take(gc, (size_t)N * HEADS * 4);
    float*    denom = (float*)ws_take(gc, (size_t)N * HEADS * 4);
    // LSTM-phase scratch (aliases GAT-phase; phases are sequential)
    char* lc = scratch;
    float* zbuf = (float*)ws_take(lc, (size_t)N * 4 * HID * 4);
    float* hst  = (float*)ws_take(lc, (size_t)N * HID * 4);
    float* cst  = (float*)ws_take(lc, (size_t)N * HID * 4);
    bf16*  hstb = (bf16*)ws_take(lc, (size_t)N * HID * 2);

    auto gemm = [&](const bf16* A, const bf16* B, const float* bias, float* C,
                    int M, int K, int Nc, int accum) {
        dim3 grid((M + BM - 1) / BM, Nc / BN);
        gemm_bf16_wmma<<<grid, 256, 0, stream>>>(A, B, bias, C, M, K, Nc, accum);
    };
    auto castw = [&](const float* s, bf16* d, long n) {
        cast_f32_bf16<<<(int)((n + 255) / 256), 256, 0, stream>>>(s, d, n);
    };

    // ---- weight prep (bf16) ----
    castw(projW, projWb, 96L * 128);
    castw(g1Wl, g1Wlb, 128L * 512);
    castw(g1Wr, g1Wrb, 128L * 512);
    castw(g2Wl, g2Wlb, 128L * 512);
    castw(g2Wr, g2Wrb, 128L * 512);
    transpose_cast<<<(512 * 128 + 255) / 256, 256, 0, stream>>>(Wih_f, WihTf, 512, 128);
    transpose_cast<<<(512 * 128 + 255) / 256, 256, 0, stream>>>(Whh_f, WhhTf, 512, 128);
    transpose_cast<<<(512 * 128 + 255) / 256, 256, 0, stream>>>(Wih_b, WihTb, 512, 128);
    transpose_cast<<<(512 * 128 + 255) / 256, 256, 0, stream>>>(Whh_b, WhhTb, 512, 128);

    const int ewaves  = Etot * HEADS;                 // one wave per (edge, head)
    const int eblocks = (ewaves + 7) / 8;             // 8 waves per 256-thread block
    const int etblk   = (ewaves + 255) / 256;

    auto gat_layer = [&](const bf16* inb, const bf16* Wlb, const bf16* Wrb,
                         const float* bl, const float* br, const float* att,
                         const float* gbias, const float* res,
                         const float* lng, const float* lnb,
                         float* outf, bf16* outb) {
        gemm(inb, Wlb, bl, xl, N, HID, HEADS * HID, 0);
        gemm(inb, Wrb, br, xr, N, HID, HEADS * HID, 0);
        fill_u32<<<2048, 256, 0, stream>>>((unsigned*)agg, 0u, (long)N * HEADS * HID);
        fill_u32<<<1024, 256, 0, stream>>>(emaxu, 0x007FFFFFu, (long)N * HEADS); // enc(-inf)
        fill_u32<<<1024, 256, 0, stream>>>((unsigned*)denom, 0u, (long)N * HEADS);
        edge_scores<<<eblocks, 256, 0, stream>>>(xl, xr, srcA, dstA, att, ev, emaxu, Etot, Eorig);
        edge_exp<<<etblk, 256, 0, stream>>>(ev, emaxu, denom, dstA, Etot, Eorig);
        edge_aggr<<<eblocks, 256, 0, stream>>>(xl, ev, denom, srcA, dstA, agg, Etot, Eorig);
        node_ln<<<N, HID, 0, stream>>>(agg, gbias, res, lng, lnb, outf, outb);
    };

    // ---- temporal GAT encoder ----
    for (int t = 0; t < T; ++t) {
        const float* x_t = x + (size_t)t * N * 64;
        concat_cast<<<(int)(((long)N * 96 + 255) / 256), 256, 0, stream>>>(x_t, emb, A96b, N);
        gemm(A96b, projWb, projb, xt, N, 96, HID, 0);
        cast_f32_bf16<<<(int)(((long)N * HID + 255) / 256), 256, 0, stream>>>(xt, nodeb, (long)N * HID);
        // layer 1: residual = xt -> hbuf (f32) + nodeb (bf16 for layer 2)
        gat_layer(nodeb, g1Wlb, g1Wrb, g1bl, g1br, g1att, g1bias, xt, ln1g, ln1b, hbuf, nodeb);
        // layer 2: residual = hbuf -> seqb[t] (bf16 only)
        gat_layer(nodeb, g2Wlb, g2Wrb, g2bl, g2br, g2att, g2bias, hbuf, ln2g, ln2b,
                  nullptr, seqb + (size_t)t * N * HID);
    }

    // ---- bidirectional LSTM ----
    const int gthr = (int)(((long)N * HID + 255) / 256);
    auto lstm_dir = [&](const bf16* WihT, const bf16* WhhT, const float* bih,
                        const float* bhh, float* y, int forward) {
        fill_u32<<<1024, 256, 0, stream>>>((unsigned*)hst, 0u, (long)N * HID);
        fill_u32<<<1024, 256, 0, stream>>>((unsigned*)cst, 0u, (long)N * HID);
        fill_u32<<<1024, 256, 0, stream>>>((unsigned*)hstb, 0u, (long)N * HID / 2);
        for (int s = 0; s < T; ++s) {
            int t = forward ? s : (T - 1 - s);
            gemm(seqb + (size_t)t * N * HID, WihT, nullptr, zbuf, N, HID, 4 * HID, 0);
            gemm(hstb, WhhT, nullptr, zbuf, N, HID, 4 * HID, 1);
            lstm_gate<<<gthr, 256, 0, stream>>>(zbuf, bih, bhh, hst, cst, hstb,
                                                y + (size_t)t * N * HID, N);
        }
    };
    lstm_dir(WihTf, WhhTf, bih_f, bhh_f, yfub, 1);
    lstm_dir(WihTb, WhhTb, bih_b, bhh_b, ybub, 0);

    // ---- attention pooling + MLP head ----
    attn_fc<<<N, 256, 0, stream>>>(yfub, ybub, attnW, attnb, fc1W, fc1b, fc2W, fc2b,
                                   (float*)d_out, N, T);
}
